// TransformerEncoderBlock_83975200572059
// MI455X (gfx1250) — matrix operand order repaired and checked by reading, not compiled, not run on hardware
//
#include <hip/hip_runtime.h>
#include <math.h>

// ---------------------------------------------------------------------------
// Problem dims
// ---------------------------------------------------------------------------
constexpr int Bsz = 4;
constexpr int S   = 1024;
constexpr int E   = 768;
constexpr int H   = 8;
constexpr int HD  = 96;
constexpr int FF  = 3072;
constexpr int NT  = Bsz * S;                    // 4096
constexpr float SCALING = 0.102062072615966f;   // 96^-0.5

typedef __attribute__((ext_vector_type(16))) _Float16 v16h;
typedef __attribute__((ext_vector_type(8)))  _Float16 v8h;
typedef __attribute__((ext_vector_type(8)))  float    v8f;
typedef unsigned int u32x4 __attribute__((ext_vector_type(4)));
typedef int          i32x4 __attribute__((ext_vector_type(4)));
typedef int          i32x8 __attribute__((ext_vector_type(8)));

#if defined(__AMDGCN__) && __has_builtin(__builtin_amdgcn_tensor_load_to_lds) && \
    __has_builtin(__builtin_amdgcn_s_wait_tensorcnt)
#define USE_TDM 1
#else
#define USE_TDM 0
#endif

// Compiler fence: TDM writes LDS behind the compiler's back; force it to
// assume memory (incl. LDS) changed so staged-fragment loads are not folded.
#define LDS_FENCE() asm volatile("" ::: "memory")

// ---------------------------------------------------------------------------
// TDM: 2D tile load Global->LDS.  D# per CDNA5 ISA 8.3/8.4:
//   group0: count=1 | lds_addr | global_addr | type=2
//   group1: data_size=2B, pad_enable, pad_interval=16dw, pad_amount=4dw
//           (32-half rows land on a 40-half LDS pitch -> conflict-free b128)
//           tensor_dim0=ld, tensor_dim1=rows, tile=32 x tile_rows, stride0=ld
// ---------------------------------------------------------------------------
#if USE_TDM
__device__ __forceinline__ void tdm_load_tile(unsigned lds_byte_off,
                                              const _Float16* g, int ld,
                                              int rows, int tile_rows) {
  unsigned long long ga = (unsigned long long)(size_t)g;
  u32x4 g0;
  g0[0] = 1u;                                   // count=1, user mode
  g0[1] = lds_byte_off;                         // lds_addr
  g0[2] = (unsigned)ga;                         // global_addr[31:0]
  g0[3] = (unsigned)(ga >> 32) | 0x80000000u;   // global_addr[56:32], type=2
  i32x8 g1;
  g1[0] = (1 << 16) | (1 << 20) | (3 << 22) | (3 << 25); // dsz=2B,pad en,16dw,+4dw
  g1[1] = (ld & 0xffff) << 16;                            // tensor_dim0 lo16
  g1[2] = (int)(((unsigned)ld >> 16) | ((unsigned)(rows & 0xffff) << 16));
  g1[3] = (int)(((unsigned)rows >> 16) | (32u << 16));    // tile_dim0 = 32
  g1[4] = tile_rows;                                      // tile_dim1
  g1[5] = ld;                                             // dim0 stride lo32
  g1[6] = 0;
  g1[7] = 0;
  i32x4 z4 = {};
#if __clang_major__ >= 23
  i32x8 z8 = {};
  __builtin_amdgcn_tensor_load_to_lds(g0, g1, z4, z4, z8, 0);
#else
  __builtin_amdgcn_tensor_load_to_lds(g0, g1, z4, z4, 0);
#endif
  LDS_FENCE();
}
#endif

// ---------------------------------------------------------------------------
// WMMA fragment loaders (16x16x32 f16, wave32 layouts per ISA 7.12.2)
// ---------------------------------------------------------------------------
__device__ __forceinline__ v16h load_a_frag(const _Float16* __restrict__ A,
                                            int lda, int m0, int k0, int lane) {
  const _Float16* p = A + (size_t)(m0 + (lane & 15)) * lda + k0 + ((lane >> 4) << 3);
  v8h lo = *(const v8h*)p;
  v8h hi = *(const v8h*)(p + 16);
  v16h r;
#pragma unroll
  for (int i = 0; i < 8; ++i) { r[i] = lo[i]; r[i + 8] = hi[i]; }
  return r;
}

__device__ __forceinline__ v16h load_b_frag(const _Float16* __restrict__ W,
                                            int ldw, int n0, int k0, int lane) {
  const _Float16* p = W + (size_t)(n0 + (lane & 15)) * ldw + k0 + ((lane >> 4) << 3);
  v8h lo = *(const v8h*)p;
  v8h hi = *(const v8h*)(p + 16);
  v16h r;
#pragma unroll
  for (int i = 0; i < 8; ++i) { r[i] = lo[i]; r[i + 8] = hi[i]; }
  return r;
}

// LDS variants: tile rows have PITCH=40 halves (80 B) to avoid bank conflicts.
__device__ __forceinline__ v16h lds_a_frag(const _Float16* sA, int r0, int lane) {
  const _Float16* p = sA + (r0 + (lane & 15)) * 40 + ((lane >> 4) << 3);
  v8h lo = *(const v8h*)p;
  v8h hi = *(const v8h*)(p + 16);
  v16h r;
#pragma unroll
  for (int i = 0; i < 8; ++i) { r[i] = lo[i]; r[i + 8] = hi[i]; }
  return r;
}

__device__ __forceinline__ v16h lds_b_frag(const _Float16* sB, int r0, int lane) {
  const _Float16* p = sB + (r0 + (lane & 15)) * 40 + ((lane >> 4) << 3);
  v8h lo = *(const v8h*)p;
  v8h hi = *(const v8h*)(p + 16);
  v16h r;
#pragma unroll
  for (int i = 0; i < 8; ++i) { r[i] = lo[i]; r[i + 8] = hi[i]; }
  return r;
}

__device__ __forceinline__ float gelu_exact(float x) {
  return 0.5f * x * (1.0f + erff(x * 0.70710678118654752f));
}

// ---------------------------------------------------------------------------
// TDM/LDS double-buffered WMMA GEMM:  out = act(A @ W^T + bias) + residual
// A:[M,K] f16, W:[N,K] f16, both row-major, lda = ldw = K.
// Block: 256 thr = 8 waves (4x2), tile 128(M) x 64(N); wave tile 32x32.
// LDS halves: A buf0 [0,5120) A buf1 [5120,10240) B buf0 [10240,12800)
//             B buf1 [12800,15360)
// ---------------------------------------------------------------------------
#define WMMA_STEP(sA_, sB_)                                                     \
  {                                                                             \
    v16h a0 = lds_a_frag(sA_, wm, lane);                                        \
    v16h a1 = lds_a_frag(sA_, wm + 16, lane);                                   \
    v16h b0 = lds_b_frag(sB_, wn, lane);                                        \
    v16h b1 = lds_b_frag(sB_, wn + 16, lane);                                   \
    acc[0][0] = __builtin_amdgcn_wmma_f32_16x16x32_f16(false, a0, false, b0,    \
                                        (short)0, acc[0][0], false, false);     \
    acc[0][1] = __builtin_amdgcn_wmma_f32_16x16x32_f16(false, a0, false, b1,    \
                                        (short)0, acc[0][1], false, false);     \
    acc[1][0] = __builtin_amdgcn_wmma_f32_16x16x32_f16(false, a1, false, b0,    \
                                        (short)0, acc[1][0], false, false);     \
    acc[1][1] = __builtin_amdgcn_wmma_f32_16x16x32_f16(false, a1, false, b1,    \
                                        (short)0, acc[1][1], false, false);     \
  }

__global__ __launch_bounds__(256) void k_gemm(
    const _Float16* __restrict__ A, const _Float16* __restrict__ W,
    const float* __restrict__ bias, const float* __restrict__ residual,
    float* __restrict__ outF, _Float16* __restrict__ outH,
    int M, int N, int K, int act) {
  __shared__ _Float16 smem[2 * 128 * 40 + 2 * 64 * 40];   // 30720 bytes
  const int lane = threadIdx.x & 31;
  const int wave = threadIdx.x >> 5;
  const int wm = (wave >> 1) * 32;          // wave row origin in block tile
  const int wn = (wave & 1) * 32;           // wave col origin in block tile
  const int rowBase = blockIdx.y * 128;
  const int colBase = blockIdx.x * 64;
  const _Float16* Ab = A + (size_t)rowBase * K;
  const _Float16* Wb = W + (size_t)colBase * K;
  const int nk = K >> 5;

  v8f acc[2][2] = {};

#if USE_TDM
  // Never taken at runtime (K > 0 always), but the compiler cannot prove it:
  // keeps the LDS global "written" so staged-fragment loads are not undef-folded.
  if (K == 0) smem[threadIdx.x] = (_Float16)0;

  const bool issuer = (wave == 0);
  if (issuer) {
    tdm_load_tile(0u,     Ab, K, 128, 128);
    tdm_load_tile(20480u, Wb, K, 64, 64);
    __builtin_amdgcn_s_wait_tensorcnt(0);
    LDS_FENCE();
  }
  __syncthreads();
  for (int kt = 0; kt < nk; ++kt) {
    const int cur = kt & 1;
    if (kt + 1 < nk && issuer) {
      const int nxt = 1 - cur;
      tdm_load_tile((unsigned)(nxt * 10240),        Ab + ((kt + 1) << 5), K, 128, 128);
      tdm_load_tile((unsigned)(20480 + nxt * 5120), Wb + ((kt + 1) << 5), K, 64, 64);
    }
    const _Float16* sA = smem + cur * 5120;
    const _Float16* sB = smem + 10240 + cur * 2560;
    WMMA_STEP(sA, sB);
    if (kt + 1 < nk) {
      if (issuer) {
        __builtin_amdgcn_s_wait_tensorcnt(0);
        LDS_FENCE();
      }
      __syncthreads();
    }
  }
#else
  // Fallback: cooperative b128 staging (single buffer).
  for (int kt = 0; kt < nk; ++kt) {
    if (kt) __syncthreads();
    {
      const int t = threadIdx.x;
      const int r = t >> 1, c = (t & 1) << 4;            // A: 16 halves/thread
      const _Float16* src = Ab + (size_t)r * K + (kt << 5) + c;
      _Float16* dst = smem + r * 40 + c;
      *(v8h*)dst = *(const v8h*)src;
      *(v8h*)(dst + 8) = *(const v8h*)(src + 8);
      const int rb = t >> 2, cb = (t & 3) << 3;          // B: 8 halves/thread
      const _Float16* srcb = Wb + (size_t)rb * K + (kt << 5) + cb;
      *(v8h*)(smem + 10240 + rb * 40 + cb) = *(const v8h*)srcb;
      if (kt + 1 < nk)
        __builtin_prefetch((const void*)(src + 32), 0, 3);
    }
    __syncthreads();
    const _Float16* sA = smem;
    const _Float16* sB = smem + 10240;
    WMMA_STEP(sA, sB);
  }
#endif

  const int colL = lane & 15;
  const int rowL = (lane >> 4) << 3;
#pragma unroll
  for (int mi = 0; mi < 2; ++mi) {
#pragma unroll
    for (int ni = 0; ni < 2; ++ni) {
      const int col = colBase + wn + ni * 16 + colL;
#pragma unroll
      for (int r = 0; r < 8; ++r) {
        const int m = rowBase + wm + mi * 16 + rowL + r;
        float v = acc[mi][ni][r];
        if (bias)     v += bias[col];
        if (act == 1) v  = gelu_exact(v);
        if (residual) v += residual[(size_t)m * N + col];
        const size_t o = (size_t)m * N + col;
        if (outF) outF[o] = v;
        if (outH) outH[o] = (_Float16)v;
      }
    }
  }
}

// ---------------------------------------------------------------------------
// Attention scores (direct-global WMMA, K=HD=96): mask-before-scale epilogue,
// written straight into the probs region of d_out.
// ---------------------------------------------------------------------------
__global__ __launch_bounds__(256) void k_scores(
    const _Float16* __restrict__ qh, const _Float16* __restrict__ kh,
    const int* __restrict__ mask, float* __restrict__ sc) {
  const int lane = threadIdx.x & 31;
  const int wave = threadIdx.x >> 5;
  const int bh = blockIdx.z;
  const int b  = bh >> 3;
  const _Float16* A = qh + (size_t)bh * S * HD;
  const _Float16* W = kh + (size_t)bh * S * HD;
  const int m0 = blockIdx.y * 64 + (wave >> 1) * 16;
  const int n0 = blockIdx.x * 32 + (wave & 1) * 16;

  v8f acc = {};
#pragma unroll
  for (int k0 = 0; k0 < HD; k0 += 32) {
    v16h a = load_a_frag(A, HD, m0, k0, lane);
    v16h bf = load_b_frag(W, HD, n0, k0, lane);
    acc = __builtin_amdgcn_wmma_f32_16x16x32_f16(false, a, false, bf,
                                                 (short)0, acc, false, false);
  }

  const int col = n0 + (lane & 15);
  const int mbase = m0 + ((lane >> 4) << 3);
  const bool masked = mask[b * S + col] != 0;
#pragma unroll
  for (int r = 0; r < 8; ++r) {
    float v = masked ? -10000.0f : acc[r];
    sc[((size_t)bh * S + (mbase + r)) * S + col] = v * SCALING;
  }
}

// ---------------------------------------------------------------------------
// Row softmax (1024 cols) fused with probs = (softmax + dist) * 0.5.
// ---------------------------------------------------------------------------
__global__ __launch_bounds__(256) void k_softmax_dist(
    float* __restrict__ sc, const float* __restrict__ dist,
    _Float16* __restrict__ ph) {
  const int t = threadIdx.x, lane = t & 31, wv = t >> 5;
  const size_t row = blockIdx.x;
  float* srow = sc + row * S;
  const float* drow = dist + row * S;
  __shared__ float red[8];

  float v[4];
#pragma unroll
  for (int i = 0; i < 4; ++i) v[i] = srow[t + 256 * i];

  float mx = fmaxf(fmaxf(v[0], v[1]), fmaxf(v[2], v[3]));
#pragma unroll
  for (int o = 16; o; o >>= 1) mx = fmaxf(mx, __shfl_xor(mx, o, 32));
  if (lane == 0) red[wv] = mx;
  __syncthreads();
  mx = red[0];
#pragma unroll
  for (int i = 1; i < 8; ++i) mx = fmaxf(mx, red[i]);
  __syncthreads();

  float sum = 0.0f;
#pragma unroll
  for (int i = 0; i < 4; ++i) { v[i] = __expf(v[i] - mx); sum += v[i]; }
#pragma unroll
  for (int o = 16; o; o >>= 1) sum += __shfl_xor(sum, o, 32);
  if (lane == 0) red[wv] = sum;
  __syncthreads();
  float tot = 0.0f;
#pragma unroll
  for (int i = 0; i < 8; ++i) tot += red[i];
  const float inv = 1.0f / tot;

#pragma unroll
  for (int i = 0; i < 4; ++i) {
    const float p = (v[i] * inv + drow[t + 256 * i]) * 0.5f;
    srow[t + 256 * i] = p;
    ph[row * S + t + 256 * i] = (_Float16)p;
  }
}

// vT[bh][d][j] = v[bh][j][d]
__global__ __launch_bounds__(256) void k_transpose_v(
    const _Float16* __restrict__ vh, _Float16* __restrict__ vT) {
  const size_t i = (size_t)blockIdx.x * 256 + threadIdx.x;
  const int d = (int)(i % HD);
  const size_t t2 = i / HD;
  const int j = (int)(t2 % S);
  const int bh = (int)(t2 / S);
  vT[((size_t)bh * HD + d) * S + j] = vh[i];
}

// ---------------------------------------------------------------------------
// ctx = probs @ v per (b,h); output scattered into [B,S,E] (head transpose).
// ---------------------------------------------------------------------------
__global__ __launch_bounds__(256) void k_ctx(
    const _Float16* __restrict__ ph, const _Float16* __restrict__ vT,
    _Float16* __restrict__ ctx) {
  const int lane = threadIdx.x & 31;
  const int wave = threadIdx.x >> 5;
  const int bh = blockIdx.z;
  const int b = bh >> 3, h = bh & 7;
  const _Float16* A = ph + (size_t)bh * S * S;
  const _Float16* W = vT + (size_t)bh * HD * S;
  const int m0 = blockIdx.y * 64 + (wave >> 1) * 16;
  const int n0 = blockIdx.x * 32 + (wave & 1) * 16;

  v8f acc = {};
  for (int k0 = 0; k0 < S; k0 += 32) {
    v16h a = load_a_frag(A, S, m0, k0, lane);
    v16h bf = load_b_frag(W, S, n0, k0, lane);
    acc = __builtin_amdgcn_wmma_f32_16x16x32_f16(false, a, false, bf,
                                                 (short)0, acc, false, false);
  }

  const int col = n0 + (lane & 15);
  const int mbase = m0 + ((lane >> 4) << 3);
#pragma unroll
  for (int r = 0; r < 8; ++r) {
    const int m = mbase + r;
    ctx[((size_t)(b * S + m)) * E + h * HD + col] = (_Float16)acc[r];
  }
}

// Fused LayerNorm (biased var, eps=1e-5) -> f16. One row / block.
__global__ __launch_bounds__(256) void k_layernorm(
    const float* __restrict__ x, const float* __restrict__ g,
    const float* __restrict__ bb, _Float16* __restrict__ out) {
  const int t = threadIdx.x, lane = t & 31, wv = t >> 5;
  const size_t row = blockIdx.x;
  const float* xr = x + row * E;
  float a0 = xr[t], a1 = xr[t + 256], a2 = xr[t + 512];
  float s = a0 + a1 + a2;
  float q = a0 * a0 + a1 * a1 + a2 * a2;
#pragma unroll
  for (int o = 16; o; o >>= 1) {
    s += __shfl_xor(s, o, 32);
    q += __shfl_xor(q, o, 32);
  }
  __shared__ float rs[8], rq[8];
  if (lane == 0) { rs[wv] = s; rq[wv] = q; }
  __syncthreads();
  float Sm = 0.0f, Qm = 0.0f;
#pragma unroll
  for (int i = 0; i < 8; ++i) { Sm += rs[i]; Qm += rq[i]; }
  const float mean = Sm * (1.0f / 768.0f);
  const float var  = Qm * (1.0f / 768.0f) - mean * mean;
  const float inv  = rsqrtf(var + 1e-5f);
  _Float16* orow = out + row * E;
  orow[t]       = (_Float16)((a0 - mean) * inv * g[t] + bb[t]);
  orow[t + 256] = (_Float16)((a1 - mean) * inv * g[t + 256] + bb[t + 256]);
  orow[t + 512] = (_Float16)((a2 - mean) * inv * g[t + 512] + bb[t + 512]);
}

__global__ __launch_bounds__(256) void k_f2h(const float* __restrict__ in,
                                             _Float16* __restrict__ out, int n) {
  const int i = blockIdx.x * 256 + threadIdx.x;
  if (i < n) out[i] = (_Float16)in[i];
}

// ---------------------------------------------------------------------------
extern "C" void kernel_launch(void* const* d_in, const int* in_sizes, int n_in,
                              void* d_out, int out_size, void* d_ws, size_t ws_size,
                              hipStream_t stream) {
  const float* x    = (const float*)d_in[0];
  const float* dist = (const float*)d_in[1];
  const int*   mask = (const int*)  d_in[2];
  const float* Wq   = (const float*)d_in[3];
  const float* bq   = (const float*)d_in[4];
  const float* Wk   = (const float*)d_in[5];
  const float* bk   = (const float*)d_in[6];
  const float* Wv   = (const float*)d_in[7];
  const float* bv   = (const float*)d_in[8];
  const float* Wo   = (const float*)d_in[9];
  const float* bo   = (const float*)d_in[10];
  const float* ln1g = (const float*)d_in[11];
  const float* ln1b = (const float*)d_in[12];
  const float* W1   = (const float*)d_in[13];
  const float* b1   = (const float*)d_in[14];
  const float* W2   = (const float*)d_in[15];
  const float* b2   = (const float*)d_in[16];
  const float* ln2g = (const float*)d_in[17];
  const float* ln2b = (const float*)d_in[18];

  float* outX  = (float*)d_out;                   // [B,S,E]
  float* probs = (float*)d_out + (size_t)NT * E;  // [B,H,S,S]

  constexpr size_t E2  = (size_t)E * E;
  constexpr size_t W1s = (size_t)FF * E;
  constexpr size_t NE  = (size_t)NT * E;
  constexpr size_t PR  = (size_t)Bsz * H * S * S;
  constexpr size_t FF1 = (size_t)NT * FF;

  _Float16* wq_h = (_Float16*)d_ws;
  _Float16* wk_h = wq_h + E2;
  _Float16* wv_h = wk_h + E2;
  _Float16* wo_h = wv_h + E2;
  _Float16* w1_h = wo_h + E2;
  _Float16* w2_h = w1_h + W1s;
  _Float16* hbf  = w2_h + W1s;
  _Float16* qbf  = hbf + NE;
  _Float16* kbf  = qbf + NE;
  _Float16* vbf  = kbf + NE;
  _Float16* vT   = vbf + NE;
  _Float16* prh  = vT + NE;
  _Float16* ctxh = prh + PR;
  _Float16* h2bf = ctxh + NE;
  _Float16* ff1h = h2bf + NE;
  float*    x1   = (float*)(ff1h + FF1);

  // weights fp32 -> f16
  k_f2h<<<(int)((E2  + 255) / 256), 256, 0, stream>>>(Wq, wq_h, (int)E2);
  k_f2h<<<(int)((E2  + 255) / 256), 256, 0, stream>>>(Wk, wk_h, (int)E2);
  k_f2h<<<(int)((E2  + 255) / 256), 256, 0, stream>>>(Wv, wv_h, (int)E2);
  k_f2h<<<(int)((E2  + 255) / 256), 256, 0, stream>>>(Wo, wo_h, (int)E2);
  k_f2h<<<(int)((W1s + 255) / 256), 256, 0, stream>>>(W1, w1_h, (int)W1s);
  k_f2h<<<(int)((W1s + 255) / 256), 256, 0, stream>>>(W2, w2_h, (int)W1s);

  // LN1
  k_layernorm<<<NT, 256, 0, stream>>>(x, ln1g, ln1b, hbf);

  // QKV projections (TDM/LDS GEMM: grid = (N/64, M/128))
  dim3 gP(E / 64, NT / 128);
  k_gemm<<<gP, 256, 0, stream>>>(hbf, wq_h, bq, nullptr, nullptr, qbf, NT, E, E, 0);
  k_gemm<<<gP, 256, 0, stream>>>(hbf, wk_h, bk, nullptr, nullptr, kbf, NT, E, E, 0);
  k_gemm<<<gP, 256, 0, stream>>>(hbf, wv_h, bv, nullptr, nullptr, vbf, NT, E, E, 0);

  // scores -> d_out probs region (staging), then fused softmax + dist bias
  k_scores<<<dim3(S / 32, S / 64, Bsz * H), 256, 0, stream>>>(qbf, kbf, mask, probs);
  k_softmax_dist<<<Bsz * H * S, 256, 0, stream>>>(probs, dist, prh);

  // ctx
  k_transpose_v<<<(int)(NE / 256), 256, 0, stream>>>(vbf, vT);
  k_ctx<<<dim3(HD / 32, S / 64, Bsz * H), 256, 0, stream>>>(prh, vT, ctxh);

  // output projection + residual
  k_gemm<<<gP, 256, 0, stream>>>(ctxh, wo_h, bo, x, x1, nullptr, NT, E, E, 0);

  // LN2 + FFN
  k_layernorm<<<NT, 256, 0, stream>>>(x1, ln2g, ln2b, h2bf);
  k_gemm<<<dim3(FF / 64, NT / 128), 256, 0, stream>>>(h2bf, w1_h, b1, nullptr,
                                                      nullptr, ff1h, NT, FF, E, 1);
  k_gemm<<<gP, 256, 0, stream>>>(ff1h, w2_h, b2, x1, outX, nullptr, NT, E, FF, 0);
}
